// Hawk_35442070127161
// MI455X (gfx1250) — compile-verified, matrix operand order checked by reading, not run on hardware
//
#include <hip/hip_runtime.h>
#include <hip/hip_bf16.h>
#include <math.h>

// ---------------------------------------------------------------------------
// Hawk / RG-LRU forward for MI455X (gfx1250, wave32, WMMA + TDM).
//   N=4, T=2048, DIM=1024, HID=1536, K=4
// Pipeline: cvt(f32->f16) -> WMMA GEMM1 -> causal depthwise conv ->
//           WMMA GEMM2 -> alpha/xi elementwise -> 3-pass chunked scan ->
//           WMMA GEMM3 -> out
// GEMM tiles are fed by the Tensor Data Mover (tensor_load_to_lds, TENSORcnt)
// ---------------------------------------------------------------------------

typedef _Float16 v16h __attribute__((ext_vector_type(16)));
typedef _Float16 v8h  __attribute__((ext_vector_type(8)));
typedef float    v8f  __attribute__((ext_vector_type(8)));
typedef unsigned int v4u __attribute__((ext_vector_type(4)));
typedef int      v4i __attribute__((ext_vector_type(4)));
typedef int      v8i __attribute__((ext_vector_type(8)));

#define NB   4
#define TT   2048
#define DIM  1024
#define HID  1536
#define ROWS (NB * TT)          // 8192
#define TWOH (2 * HID)          // 3072

#define BM 128
#define BN 128
#define BK 32
#define LDP 40                  // padded LDS row stride in f16 (80B, 16B-aligned)

__device__ __forceinline__ float gelu_exact(float x) {
    return 0.5f * x * (1.0f + erff(x * 0.7071067811865476f));
}
__device__ __forceinline__ float sigmoidf_(float x) {
    return 1.0f / (1.0f + expf(-x));
}

// ----------------------- TDM 2D tile load (global->LDS) --------------------
// Loads a BM(rows) x BK(f16) tile from a row-major [rows, elemsK] f16 tensor
// into LDS with hardware padding: 64B of data then 16B pad per row, i.e. the
// LDS row stride is LDP=40 f16 — exactly the layout the WMMA fragments read.
//   pad_interval=3 -> pad after 16 DWORDs (64B); pad_amount=3 -> 4 DWORDs (16B)
__device__ __forceinline__ void tdm_load_tile(const _Float16* gptr, void* lptr,
                                              int elemsK, int rows) {
    unsigned long long ga = (unsigned long long)(uintptr_t)gptr;
    unsigned int lds = (unsigned int)(uintptr_t)lptr;  // low 32b = LDS offset
    v4u g0;
    g0[0] = 1u;                                        // count=1 (valid D#)
    g0[1] = lds;                                       // lds_addr
    g0[2] = (unsigned int)(ga & 0xFFFFFFFFu);          // global_addr[31:0]
    g0[3] = (unsigned int)((ga >> 32) & 0x01FFFFFFu)   // global_addr[56:32]
            | (2u << 30);                              // type=2 ("image")
    v8i g1;
    g1[0] = (int)((1u << 16)        // data_size = 1 -> 2 bytes
                  | (1u << 20)      // pad_enable
                  | (3u << 22)      // pad_interval: 16 DWORDs
                  | (3u << 25));    // pad_amount: 4 DWORDs
    g1[1] = (int)(((unsigned int)elemsK & 0xFFFFu) << 16);          // t_dim0 lo
    g1[2] = (int)((((unsigned int)elemsK >> 16) & 0xFFFFu)          // t_dim0 hi
                  | (((unsigned int)rows & 0xFFFFu) << 16));        // t_dim1 lo
    g1[3] = (int)((((unsigned int)rows >> 16) & 0xFFFFu)            // t_dim1 hi
                  | ((unsigned int)BK << 16));                      // tile_dim0
    g1[4] = BM;                     // tile_dim1 = 128, tile_dim2 = 0
    g1[5] = elemsK;                 // tensor_dim0_stride[31:0] (element units)
    g1[6] = 0;                      // stride hi / tensor_dim1_stride lo
    g1[7] = 0;
    v4i z4 = {0, 0, 0, 0};
#if defined(__clang_major__) && (__clang_major__ >= 23)
    v8i z8 = {0, 0, 0, 0, 0, 0, 0, 0};
    __builtin_amdgcn_tensor_load_to_lds(g0, g1, z4, z4, z8, 0);
#else
    __builtin_amdgcn_tensor_load_to_lds(g0, g1, z4, z4, 0);
#endif
}

// ------------------------------- f32 -> f16 --------------------------------
__global__ void cvt_f32_to_f16(const float* __restrict__ s,
                               _Float16* __restrict__ d, int n) {
    int i = blockIdx.x * blockDim.x + threadIdx.x;
    if (i < n) d[i] = (_Float16)s[i];
}

// ------------------------------- WMMA GEMM ---------------------------------
// C[ROWS, Nc] = A[ROWS, K] * B[Nc, K]^T  (both K-major / "NT" layout),
// f16 operands, f32 accumulate, TDM-fed double-buffered LDS tiles.
//   EPI=0 : col <  HID -> outH[row*HID+col]      = f16(gelu(acc))   (gate)
//           col >= HID -> outF[row*HID+col-HID]  = acc              (xh raw)
//   EPI=1 : outF[row*Nc+col] = acc + bias[col]                      (fg)
//   EPI=2 : outF[row*Nc+col] = acc                                  (out)
template <int EPI>
__global__ __launch_bounds__(256) void gemm_wmma(
    const _Float16* __restrict__ A, const _Float16* __restrict__ B,
    int Nc, int K,
    float* __restrict__ outF, _Float16* __restrict__ outH,
    const float* __restrict__ bias) {
    __shared__ __align__(16) _Float16 lA[2][BM][LDP];
    __shared__ __align__(16) _Float16 lB[2][BN][LDP];

    const int tid  = threadIdx.x;
    const int wave = tid >> 5;
    const int lane = tid & 31;
    const int rowBase = blockIdx.y * BM;
    const int colBase = blockIdx.x * BN;
    const int wm = wave >> 2;   // 0..1 -> 64-row slab
    const int wn = wave & 3;    // 0..3 -> 32-col slab
    const bool issuer = (wave == 0);   // one wave drives the TDM

    v8f acc[4][2];
#pragma unroll
    for (int i = 0; i < 4; ++i)
#pragma unroll
        for (int j = 0; j < 2; ++j)
#pragma unroll
            for (int e = 0; e < 8; ++e) acc[i][j][e] = 0.0f;

    const _Float16* gAblk = A + (size_t)rowBase * K;   // block's A rows
    const _Float16* gBblk = B + (size_t)colBase * K;   // block's B rows
    const int nk = K / BK;

    // Kick off buffer 0 via the Tensor Data Mover.
    if (issuer) {
        tdm_load_tile(gAblk, &lA[0][0][0], K, BM);
        tdm_load_tile(gBblk, &lB[0][0][0], K, BN);
        __builtin_amdgcn_s_wait_tensorcnt((short)0);
    }
    __syncthreads();

    const int koff = (lane >> 4) * 8;  // lanes 0-15: K{0..7,16..23}; 16-31: +8
    const int mrow = lane & 15;
    const int pfRow = tid >> 1;        // per-thread row for L2 prefetch

    for (int kk = 0; kk < nk; ++kk) {
        const int cur = kk & 1;
        const int nxt = cur ^ 1;
        if (kk + 1 < nk) {
            if (issuer) {
                tdm_load_tile(gAblk + (size_t)(kk + 1) * BK, &lA[nxt][0][0], K, BM);
                tdm_load_tile(gBblk + (size_t)(kk + 1) * BK, &lB[nxt][0][0], K, BN);
            }
            if (kk + 2 < nk) {  // warm L2 for the tile after next (per row)
                __builtin_prefetch(
                    (const void*)(gAblk + (size_t)pfRow * K + (size_t)(kk + 2) * BK), 0, 1);
                __builtin_prefetch(
                    (const void*)(gBblk + (size_t)pfRow * K + (size_t)(kk + 2) * BK), 0, 1);
            }
        }

        // Build fragments from LDS per documented 16-bit A layout.
        v16h af[4], bf[2];
#pragma unroll
        for (int i = 0; i < 4; ++i) {
            const _Float16* p = &lA[cur][wm * 64 + i * 16 + mrow][koff];
            v8h lo = *(const v8h*)p;
            v8h hi = *(const v8h*)(p + 16);
            af[i] = __builtin_shufflevector(lo, hi, 0, 1, 2, 3, 4, 5, 6, 7,
                                            8, 9, 10, 11, 12, 13, 14, 15);
        }
#pragma unroll
        for (int j = 0; j < 2; ++j) {
            const _Float16* p = &lB[cur][wn * 32 + j * 16 + mrow][koff];
            v8h lo = *(const v8h*)p;
            v8h hi = *(const v8h*)(p + 16);
            bf[j] = __builtin_shufflevector(lo, hi, 0, 1, 2, 3, 4, 5, 6, 7,
                                            8, 9, 10, 11, 12, 13, 14, 15);
        }

#pragma unroll
        for (int i = 0; i < 4; ++i)
#pragma unroll
            for (int j = 0; j < 2; ++j)
                acc[i][j] = __builtin_amdgcn_wmma_f32_16x16x32_f16(
                    false, af[i], false, bf[j], (short)0, acc[i][j],
                    false, false);

        if (issuer) __builtin_amdgcn_s_wait_tensorcnt((short)0);
        __syncthreads();
    }

    // Epilogue. C/D layout: VGPR e -> M = e + 8*(lane/16), N = lane%16.
    const int msel = lane >> 4;
#pragma unroll
    for (int i = 0; i < 4; ++i)
#pragma unroll
        for (int j = 0; j < 2; ++j) {
            const int col = colBase + wn * 32 + j * 16 + (lane & 15);
#pragma unroll
            for (int e = 0; e < 8; ++e) {
                const int row = rowBase + wm * 64 + i * 16 + e + 8 * msel;
                const float v = acc[i][j][e];
                if (EPI == 0) {
                    if (col < HID)
                        outH[(size_t)row * HID + col] = (_Float16)gelu_exact(v);
                    else
                        outF[(size_t)row * HID + (col - HID)] = v;
                } else if (EPI == 1) {
                    outF[(size_t)row * Nc + col] = v + bias[col];
                } else {
                    outF[(size_t)row * Nc + col] = v;
                }
            }
        }
}

// --------------------------- causal depthwise conv -------------------------
__global__ void conv_kernel(const float* __restrict__ xraw,
                            const float* __restrict__ cw,
                            const float* __restrict__ cb,
                            float* __restrict__ xc,
                            _Float16* __restrict__ xc16, int total) {
    int idx = blockIdx.x * blockDim.x + threadIdx.x;
    if (idx >= total) return;
    const int row = idx / HID;
    const int h   = idx - row * HID;
    const int n   = row >> 11;       // /TT
    const int t   = row & (TT - 1);
    float acc = cb[h];
#pragma unroll
    for (int k = 0; k < 4; ++k) {
        const int tt = t - 3 + k;
        if (tt >= 0)
            acc += xraw[((size_t)(n << 11) + tt) * HID + h] * cw[h * 4 + k];
    }
    xc[idx]   = acc;
    xc16[idx] = (_Float16)acc;
}

// ------------------------------ alpha / xi ---------------------------------
__global__ void alphaxi_kernel(const float* __restrict__ fg,
                               const float* __restrict__ fb,
                               const float* __restrict__ xc,
                               float* __restrict__ alpha,
                               float* __restrict__ xi, int total) {
    int idx = blockIdx.x * blockDim.x + threadIdx.x;
    if (idx >= total) return;
    const int row = idx / HID;
    const int h   = idx - row * HID;
    const float forget = fg[(size_t)row * TWOH + h];
    const float inp    = fg[(size_t)row * TWOH + HID + h];
    const float sp = log1pf(expf(fb[h]));            // softplus(forget_base)
    const float la = -8.0f * sp * sigmoidf_(forget); // log_alpha
    const float a  = expf(la);
    const float beta = sqrtf(fmaxf(0.0f, 1.0f - a * a));
    alpha[idx] = a;
    xi[idx]    = beta * sigmoidf_(inp) * xc[idx];
}

// --------------------------- chunked linear scan ---------------------------
#define NCH 16
#define TCH (TT / NCH)   // 128

__global__ void scan_pass1(const float* __restrict__ alpha,
                           const float* __restrict__ xi,
                           float* __restrict__ Ac, float* __restrict__ Hc) {
    int idx = blockIdx.x * blockDim.x + threadIdx.x;   // (n, chunk, h)
    if (idx >= NB * NCH * HID) return;
    const int h = idx % HID;
    const int r = idx / HID;
    const int c = r % NCH;
    const int n = r / NCH;
    float a = 1.0f, hh = 0.0f;
    size_t base = ((size_t)n * TT + c * TCH) * HID + h;
    for (int t = 0; t < TCH; ++t) {
        const float al = alpha[base];
        hh = al * hh + xi[base];
        a *= al;
        base += HID;
    }
    Ac[idx] = a;
    Hc[idx] = hh;
}

__global__ void scan_pass2(const float* __restrict__ Ac,
                           const float* __restrict__ Hc,
                           float* __restrict__ carry) {
    int idx = blockIdx.x * blockDim.x + threadIdx.x;   // (n, h)
    if (idx >= NB * HID) return;
    const int h = idx % HID;
    const int n = idx / HID;
    float cv = 0.0f;
    for (int c = 0; c < NCH; ++c) {
        const size_t j = ((size_t)n * NCH + c) * HID + h;
        carry[j] = cv;
        cv = Ac[j] * cv + Hc[j];
    }
}

__global__ void scan_pass3(const float* __restrict__ alpha,
                           const float* __restrict__ xi,
                           const float* __restrict__ carry,
                           const _Float16* __restrict__ gg,
                           _Float16* __restrict__ z16) {
    int idx = blockIdx.x * blockDim.x + threadIdx.x;   // (n, chunk, h)
    if (idx >= NB * NCH * HID) return;
    const int h = idx % HID;
    const int r = idx / HID;
    const int c = r % NCH;
    const int n = r / NCH;
    float hh = carry[idx];
    size_t base = ((size_t)n * TT + c * TCH) * HID + h;
    for (int t = 0; t < TCH; ++t) {
        hh = alpha[base] * hh + xi[base];
        z16[base] = (_Float16)((float)gg[base] * hh);  // gelu(gate) * h
        base += HID;
    }
}

// ------------------------------- launcher ----------------------------------
extern "C" void kernel_launch(void* const* d_in, const int* in_sizes, int n_in,
                              void* d_out, int out_size, void* d_ws,
                              size_t ws_size, hipStream_t stream) {
    const float* x    = (const float*)d_in[0];  // [N,T,DIM]
    const float* w1   = (const float*)d_in[1];  // [2H, DIM]
    const float* cw   = (const float*)d_in[2];  // [H,1,4]
    const float* cb   = (const float*)d_in[3];  // [H]
    const float* w2   = (const float*)d_in[4];  // [2H, H]
    const float* gb   = (const float*)d_in[5];  // [2H]
    const float* fb   = (const float*)d_in[6];  // [H]
    const float* w3   = (const float*)d_in[7];  // [DIM, H]
    float* out = (float*)d_out;                 // [N,T,DIM]

    char* ws = (char*)d_ws;
    size_t off = 0;
    auto carve = [&](size_t bytes) -> char* {
        char* p = ws + off;
        off += (bytes + 255) & ~(size_t)255;
        return p;
    };
    _Float16* x16  = (_Float16*)carve((size_t)ROWS * DIM * 2);
    _Float16* w1h  = (_Float16*)carve((size_t)TWOH * DIM * 2);
    _Float16* w2h  = (_Float16*)carve((size_t)TWOH * HID * 2);
    _Float16* w3h  = (_Float16*)carve((size_t)DIM * HID * 2);
    _Float16* gg   = (_Float16*)carve((size_t)ROWS * HID * 2);   // gelu(gate)
    float*    xraw = (float*)   carve((size_t)ROWS * HID * 4);   // xh pre-conv
    float*    xc   = (float*)   carve((size_t)ROWS * HID * 4);   // xh post-conv
    _Float16* xc16 = (_Float16*)carve((size_t)ROWS * HID * 2);
    float*    fg   = (float*)   carve((size_t)ROWS * TWOH * 4);
    float*    alp  = (float*)   carve((size_t)ROWS * HID * 4);
    float*    xi   = (float*)   carve((size_t)ROWS * HID * 4);
    _Float16* z16  = (_Float16*)carve((size_t)ROWS * HID * 2);
    float*    Ac   = (float*)   carve((size_t)NB * NCH * HID * 4);
    float*    Hc   = (float*)   carve((size_t)NB * NCH * HID * 4);
    float*    cry  = (float*)   carve((size_t)NB * NCH * HID * 4);
    (void)ws_size; (void)in_sizes; (void)n_in; (void)out_size;

    const int TPB = 256;
    // 1) f32 -> f16 conversions
    cvt_f32_to_f16<<<(ROWS * DIM + TPB - 1) / TPB, TPB, 0, stream>>>(x, x16, ROWS * DIM);
    cvt_f32_to_f16<<<(TWOH * DIM + TPB - 1) / TPB, TPB, 0, stream>>>(w1, w1h, TWOH * DIM);
    cvt_f32_to_f16<<<(TWOH * HID + TPB - 1) / TPB, TPB, 0, stream>>>(w2, w2h, TWOH * HID);
    cvt_f32_to_f16<<<(DIM * HID + TPB - 1) / TPB, TPB, 0, stream>>>(w3, w3h, DIM * HID);

    // 2) GEMM1: gi = x @ W1^T ; epilogue splits gate->gelu(f16), xh->f32
    gemm_wmma<0><<<dim3(TWOH / BN, ROWS / BM), TPB, 0, stream>>>(
        x16, w1h, TWOH, DIM, xraw, gg, nullptr);

    // 3) causal depthwise conv (+bias)
    conv_kernel<<<(ROWS * HID + TPB - 1) / TPB, TPB, 0, stream>>>(
        xraw, cw, cb, xc, xc16, ROWS * HID);

    // 4) GEMM2: fg = xh @ W2^T + gates_b
    gemm_wmma<1><<<dim3(TWOH / BN, ROWS / BM), TPB, 0, stream>>>(
        xc16, w2h, TWOH, HID, fg, nullptr, gb);

    // 5) alpha / xi elementwise
    alphaxi_kernel<<<(ROWS * HID + TPB - 1) / TPB, TPB, 0, stream>>>(
        fg, fb, xc, alp, xi, ROWS * HID);

    // 6) chunked linear recurrence + fuse gelu(gate)*h -> z16
    scan_pass1<<<(NB * NCH * HID + TPB - 1) / TPB, TPB, 0, stream>>>(alp, xi, Ac, Hc);
    scan_pass2<<<(NB * HID + TPB - 1) / TPB, TPB, 0, stream>>>(Ac, Hc, cry);
    scan_pass3<<<(NB * NCH * HID + TPB - 1) / TPB, TPB, 0, stream>>>(alp, xi, cry, gg, z16);

    // 7) GEMM3: out = z @ W3^T
    gemm_wmma<2><<<dim3(DIM / BN, ROWS / BM), TPB, 0, stream>>>(
        z16, w3h, DIM, HID, out, nullptr, nullptr);
}
